// KMeans_82360292868720
// MI455X (gfx1250) — matrix operand order repaired and checked by reading, not compiled, not run on hardware
//
#include <hip/hip_runtime.h>
#include <hip/hip_bf16.h>
#include <math.h>

// k-means assignment: N x 64 fp32 points vs 256 x 64 fp32 codebook.
// d2 = ||x||^2 + ||c||^2 - 2 x.c ; x.c via v_wmma_f32_16x16x32_f16 (f32 acc).
// Outputs: d_out[0..N)   = argmin index (stored as float value)
//          d_out[N..2N)  = min Euclidean distance (sqrt applied after min).

typedef __attribute__((ext_vector_type(16))) _Float16 v16h;
typedef __attribute__((ext_vector_type(8)))  _Float16 v8h;
typedef __attribute__((ext_vector_type(2)))  _Float16 v2h;
typedef __attribute__((ext_vector_type(8)))  float    v8f;
typedef __attribute__((ext_vector_type(4)))  float    v4f;

#define KCODES 256
#define WDIM   64
#define CB_STRIDE 72               // halves per LDS codebook row (36 dwords, padded)
#define TILES_PER_WAVE 8
#define ROWS_PER_BLOCK 1024        // 8 waves * 8 tiles * 16 rows

// Load one column-tile's B fragments (32x16 f16 ISA layout) from LDS.
__device__ __forceinline__ void load_bfrag(const _Float16* __restrict__ cb_lds,
                                           int ct, int m, int half,
                                           v16h& b0, v16h& b1)
{
    const _Float16* brow = cb_lds + (ct * 16 + m) * CB_STRIDE;
    v8h b0lo = *(const v8h*)(brow + half * 8);
    v8h b0hi = *(const v8h*)(brow + 16 + half * 8);
    v8h b1lo = *(const v8h*)(brow + 32 + half * 8);
    v8h b1hi = *(const v8h*)(brow + 32 + 16 + half * 8);
    #pragma unroll
    for (int j = 0; j < 8; ++j) {
        b0[j] = b0lo[j]; b0[8 + j] = b0hi[j];
        b1[j] = b1lo[j]; b1[8 + j] = b1hi[j];
    }
}

__global__ __launch_bounds__(256)
void kmeans_assign_wmma(const float* __restrict__ X,
                        const float* __restrict__ CB,
                        float* __restrict__ out_idx,
                        float* __restrict__ out_dist,
                        int n)
{
    __shared__ _Float16 cb_lds[KCODES * CB_STRIDE];   // 36,864 B
    __shared__ float    c2_lds[KCODES];               //  1,024 B

    const int tid = threadIdx.x;

    // ---- Stage codebook: fp32 -> f16 into LDS, and c^2. One code per thread.
    {
        const int code = tid;                 // 256 threads == 256 codes
        const float* src = CB + code * WDIM;
        _Float16* dst = &cb_lds[code * CB_STRIDE];
        float ss = 0.0f;
        #pragma unroll
        for (int i = 0; i < WDIM / 4; ++i) {
            v4f v = *(const v4f*)(src + i * 4);
            ss += v.x * v.x + v.y * v.y + v.z * v.z + v.w * v.w;
            v2h p0; p0[0] = (_Float16)v.x; p0[1] = (_Float16)v.y;
            v2h p1; p1[0] = (_Float16)v.z; p1[1] = (_Float16)v.w;
            *(v2h*)(dst + i * 4 + 0) = p0;
            *(v2h*)(dst + i * 4 + 2) = p1;
        }
        c2_lds[code] = ss;
    }
    __syncthreads();

    const int wave  = tid >> 5;
    const int lane  = tid & 31;
    const int m     = lane & 15;      // row-in-tile for A, column (code) for B/C
    const int half  = lane >> 4;      // lane group 0/1
    const int vbase = half * 8;       // C-matrix rows held by this lane: vbase..vbase+7

    const long long block_row0 = (long long)blockIdx.x * ROWS_PER_BLOCK;

    for (int t = 0; t < TILES_PER_WAVE; ++t) {
        const long long row0 = block_row0 + (long long)(t * 8 + wave) * 16;
        if (row0 >= n) continue;                       // wave-uniform: EXEC stays full

        // Clamp row so EXEC remains all-ones even for generic n (n % 16 == 0 here).
        long long xr = row0 + m; if (xr >= n) xr = n - 1;
        const float* xrow = X + xr * WDIM;

        // Prefetch next tile's rows of X (streamed, 23.3 TB/s path).
        {
            const long long nrow0 = row0 + 128;
            if (nrow0 + m < n)
                __builtin_prefetch(X + (nrow0 + m) * WDIM, 0, 0);
        }

        // ---- Build A fragments (16x32 f16, ISA lane layout) + ||x||^2 partial.
        v16h a0, a1;
        float ss = 0.0f;
        #pragma unroll
        for (int g = 0; g < 4; ++g) {
            const int ks  = g >> 1;                        // 0: K 0-31, 1: K 32-63
            const int off = ks * 32 + (g & 1) * 16 + half * 8;
            v4f u = *(const v4f*)(xrow + off);
            v4f w = *(const v4f*)(xrow + off + 4);
            ss += u.x * u.x + u.y * u.y + u.z * u.z + u.w * u.w;
            ss += w.x * w.x + w.y * w.y + w.z * w.z + w.w * w.w;
            _Float16 h[8];
            h[0] = (_Float16)u.x; h[1] = (_Float16)u.y;
            h[2] = (_Float16)u.z; h[3] = (_Float16)u.w;
            h[4] = (_Float16)w.x; h[5] = (_Float16)w.y;
            h[6] = (_Float16)w.z; h[7] = (_Float16)w.w;
            #pragma unroll
            for (int j = 0; j < 8; ++j) {
                const int e = (g & 1) * 8 + j;
                if (ks == 0) a0[e] = h[j]; else a1[e] = h[j];
            }
        }
        // Full ||x||^2 of row m: lanes m and m+16 each hold half the dims.
        const float x2_full = ss + __shfl_xor(ss, 16, 32);
        // x2 for the 8 C-matrix rows this lane owns (source lanes vbase..vbase+7).
        float x2s[8];
        #pragma unroll
        for (int v = 0; v < 8; ++v) x2s[v] = __shfl(x2_full, vbase + v, 32);

        float minv[8];
        int   mini[8];
        #pragma unroll
        for (int v = 0; v < 8; ++v) { minv[v] = INFINITY; mini[v] = 0; }

        // ---- Sweep 16 column tiles of 16 codes; B fragments software-pipelined
        //      (next tile's ds_load_b128s issue before current tile's WMMAs, so
        //      the dscnt wait before a WMMA covers loads a full iteration old).
        v16h b0_cur, b1_cur;
        load_bfrag(cb_lds, 0, m, half, b0_cur, b1_cur);

        #pragma unroll 2
        for (int ct = 0; ct < 16; ++ct) {
            v16h b0_nxt, b1_nxt;
            if (ct + 1 < 16)
                load_bfrag(cb_lds, ct + 1, m, half, b0_nxt, b1_nxt);

            v8f acc = {};
            acc = __builtin_amdgcn_wmma_f32_16x16x32_f16(
                      false, a0, false, b0_cur, (short)0, acc, false, false);
            acc = __builtin_amdgcn_wmma_f32_16x16x32_f16(
                      false, a1, false, b1_cur, (short)0, acc, false, false);

            const float c2v  = c2_lds[ct * 16 + m];
            const int   code = ct * 16 + m;
            #pragma unroll
            for (int v = 0; v < 8; ++v) {
                const float d2 = x2s[v] + c2v - 2.0f * acc[v];
                if (d2 < minv[v]) { minv[v] = d2; mini[v] = code; }
            }

            b0_cur = b0_nxt; b1_cur = b1_nxt;   // renamed away by unroll-2
        }

        // ---- Reduce (min, argmin) across the 16 code-columns (lanes) per half.
        #pragma unroll
        for (int s = 1; s <= 8; s <<= 1) {
            #pragma unroll
            for (int v = 0; v < 8; ++v) {
                const float ov = __shfl_xor(minv[v], s, 32);
                const int   oi = __shfl_xor(mini[v], s, 32);
                const bool take = (ov < minv[v]) || (ov == minv[v] && oi < mini[v]);
                minv[v] = take ? ov : minv[v];
                mini[v] = take ? oi : mini[v];
            }
        }

        // ---- Write: lanes 0-7 -> rows 0-7, lanes 16-23 -> rows 8-15.
        if ((lane & 15) < 8) {
            const int v = lane & 7;
            const long long r = row0 + vbase + v;
            if (r < n) {
                out_idx[r]  = (float)mini[v];
                out_dist[r] = sqrtf(fmaxf(minv[v], 0.0f));
            }
        }
    }
}

extern "C" void kernel_launch(void* const* d_in, const int* in_sizes, int n_in,
                              void* d_out, int out_size, void* d_ws, size_t ws_size,
                              hipStream_t stream)
{
    const float* X  = (const float*)d_in[0];   // [N, 64] fp32
    const float* CB = (const float*)d_in[1];   // [256, 64] fp32
    const int n = in_sizes[0] / WDIM;          // N = 1,000,000

    float* out      = (float*)d_out;
    float* out_idx  = out;                     // argmin as float values
    float* out_dist = out + n;                 // min distances

    const int grid = (n + ROWS_PER_BLOCK - 1) / ROWS_PER_BLOCK;
    kmeans_assign_wmma<<<grid, 256, 0, stream>>>(X, CB, out_idx, out_dist, n);
}
